// SymmetricContraction_88046829568803
// MI455X (gfx1250) — matrix operand order repaired and checked by reading, not compiled, not run on hardware
//
#include <hip/hip_runtime.h>

// ---- CDNA5 WMMA types ----
typedef __attribute__((ext_vector_type(16))) __bf16 v16bf;
typedef __attribute__((ext_vector_type(8)))  float  v8f;

#define MULC   64     // channels per node
#define NC     9      // irrep components per channel
#define KPAD   256    // padded K (216 deg-3 + 10 deg-2 + pad)
#define NCOL   729    // m*81 + i*9 + j
#define NCOLP  736    // padded to 46 tiles of 16
#define TSST   744    // LDS row stride for T (avoid bank alignment)

static __device__ __forceinline__ unsigned short f2bf(float f) {
  union { float f; unsigned int u; } c; c.f = f;
  unsigned int u = c.u;
  return (unsigned short)((u + 0x7FFFu + ((u >> 16) & 1u)) >> 16); // RNE
}

union FragBF {            // one WMMA 16x16x32 bf16 operand (per-lane 16 bf16)
  v16bf v;
  uint4 q[2];
};

// Build shared B matrix: Bm[col=(m,i,j)][k] bf16, k<216 -> U3[m,i,j,l,a] (k=l*24+a),
// 216<=k<226 -> U2[m,i,j,k-216], else 0. Stored col-major over K (row stride KPAD).
__global__ void prep_B_kernel(const float* __restrict__ U2,
                              const float* __restrict__ U3,
                              unsigned short* __restrict__ gB) {
  int c = blockIdx.x;      // 0..735
  int k = threadIdx.x;     // 0..255
  float v = 0.f;
  if (c < NCOL) {
    int m = c / 81, r = c % 81, i = r / 9, j = r % 9;
    if (k < 216) {
      int l = k / 24, a = k % 24;
      v = U3[((((m * 9 + i) * 9 + j) * 9 + l) * 24) + a];
    } else if (k < 226) {
      v = U2[(((m * 9 + i) * 9 + j) * 10) + (k - 216)];
    }
  }
  gB[c * KPAD + k] = f2bf(v);
}

__global__ void __launch_bounds__(256)
symcon_main_kernel(const float* __restrict__ x, const int* __restrict__ indices,
                   const float* __restrict__ W1, const float* __restrict__ W2,
                   const float* __restrict__ W3, const float* __restrict__ U1,
                   const unsigned short* __restrict__ gB,
                   float* __restrict__ out) {
  __shared__ float xs[MULC * NC];            // 2.3 KB  (xc, cni layout)
  __shared__ float w1s[3 * MULC];            // 768 B
  __shared__ float w2s[10 * MULC];           // 2.5 KB
  __shared__ float w3s[24 * MULC];           // 6 KB
  __shared__ float u1s[9 * 9 * 3];           // 972 B
  __shared__ unsigned short As[MULC * KPAD]; // 32 KB   (bf16 A operand)
  __shared__ float Ts[MULC * TSST];          // ~186 KB (f32 GEMM result)

  const int b   = blockIdx.x;
  const int tid = threadIdx.x;
  const int idx = indices[b];

  // ---- gather x (mul_ir -> cni) and per-element weights into LDS ----
  for (int t = tid; t < 576; t += 256) {
    float v = x[b * 576 + t];
    int u, n;
    if (t < 64)       { u = t;               n = 0; }
    else if (t < 256) { int r = t - 64;  u = r / 3; n = 1 + r % 3; }
    else              { int r = t - 256; u = r / 5; n = 4 + r % 5; }
    xs[u * NC + n] = v;
  }
  for (int t = tid; t < 24 * 64; t += 256) w3s[t] = W3[idx * 24 * 64 + t];
  for (int t = tid; t < 10 * 64; t += 256) w2s[t] = W2[idx * 10 * 64 + t];
  for (int t = tid; t < 3 * 64;  t += 256) w1s[t] = W1[idx * 3 * 64 + t];
  for (int t = tid; t < 243;     t += 256) u1s[t] = U1[t];
  __syncthreads();

  // ---- build A rows: A[u][k] = x[u,l]*w3[a,u] (k=l*24+a) | w2[k-216,u] | 0 ----
  {
    const int k = tid;                  // exactly KPAD threads
    const bool isU3 = (k < 216);
    const bool isU2 = (k >= 216) && (k < 226);
    int l = 0, a = 0;
    if (isU3) { l = k / 24; a = k % 24; }
    for (int u = 0; u < MULC; ++u) {
      float v = 0.f;
      if (isU3)      v = xs[u * NC + l] * w3s[a * 64 + u];
      else if (isU2) v = w2s[(k - 216) * 64 + u];
      As[u * KPAD + k] = f2bf(v);
    }
  }
  __syncthreads();

  // ---- GEMM: Ts[64 x 736] = A[64 x 256] x Bm[256 x 736] via v_wmma_f32_16x16x32_bf16
  const int wid    = tid >> 5;
  const int lane   = tid & 31;
  const int laneLo = lane & 15;
  const int laneHi = lane >> 4;

  for (int ct = wid; ct < NCOLP / 16; ct += 8) {   // wave-uniform loop
    // Load all 8 K-step B fragments for this column tile (kept in VGPRs).
    // B layout (this code's contiguous k order, not the ISA table order): lane n -> col n; lanes 0-15 K 0..15,
    // lanes 16-31 K 16..31 (contiguous 16 bf16 per lane).
    FragBF bf[8];
    {
      const unsigned short* bp = gB + (ct * 16 + laneLo) * KPAD + laneHi * 16;
      #pragma unroll
      for (int kk = 0; kk < 8; ++kk) {
        bf[kk].q[0] = *(const uint4*)(bp + kk * 32);
        bf[kk].q[1] = *(const uint4*)(bp + kk * 32 + 8);
      }
    }
    #pragma unroll
    for (int rt = 0; rt < 4; ++rt) {
      v8f acc = {0.f, 0.f, 0.f, 0.f, 0.f, 0.f, 0.f, 0.f};
      // A layout: lane -> row; lanes 0-15 hold
      // K 0..15, lanes 16-31 hold K 16..31 (contiguous, matching B).
      const unsigned short* ap = As + (rt * 16 + laneLo) * KPAD + laneHi * 16;
      #pragma unroll
      for (int kk = 0; kk < 8; ++kk) {
        FragBF af;
        af.q[0] = *(const uint4*)(ap + kk * 32);
        af.q[1] = *(const uint4*)(ap + kk * 32 + 8);
        acc = __builtin_amdgcn_wmma_f32_16x16x32_bf16(
            false, af.v, false, bf[kk].v, (short)0, acc, false, false);
      }
      // C/D layout: VGPR r, lanes 0-15 -> (M=r, N=lane); lanes 16-31 -> (M=8+r)
      const int crow = rt * 16 + laneHi * 8;
      const int ccol = ct * 16 + laneLo;
      #pragma unroll
      for (int r = 0; r < 8; ++r) Ts[(crow + r) * TSST + ccol] = acc[r];
    }
  }
  __syncthreads();

  // ---- fused epilogue: out[u,m] = sum_i ( U1w1[u,m,i] + sum_j T[u,m,i,j] x_j ) x_i
  for (int p = tid; p < 576; p += 256) {
    const int u = p / 9, m = p % 9;
    const float* Tu = Ts + u * TSST + m * 81;
    float accm = 0.f;
    #pragma unroll
    for (int i = 0; i < 9; ++i) {
      const float* uv = u1s + (m * 9 + i) * 3;
      float s = uv[0] * w1s[u] + uv[1] * w1s[64 + u] + uv[2] * w1s[128 + u];
      #pragma unroll
      for (int j = 0; j < 9; ++j) s += Tu[i * 9 + j] * xs[u * NC + j];
      accm += s * xs[u * NC + i];
    }
    int o;
    if (m == 0)     o = b * 576 + u;                       // l=0 block
    else if (m < 4) o = b * 576 + 64 + u * 3 + (m - 1);    // l=1 block
    else            o = b * 576 + 256 + u * 5 + (m - 4);   // l=2 block
    out[o] = accm;
  }
}

extern "C" void kernel_launch(void* const* d_in, const int* in_sizes, int n_in,
                              void* d_out, int out_size, void* d_ws, size_t ws_size,
                              hipStream_t stream) {
  const float* x   = (const float*)d_in[0];
  const int*   idx = (const int*)  d_in[1];
  const float* W1  = (const float*)d_in[2];
  const float* W2  = (const float*)d_in[3];
  const float* W3  = (const float*)d_in[4];
  const float* U1  = (const float*)d_in[5];
  const float* U2  = (const float*)d_in[6];
  const float* U3  = (const float*)d_in[7];
  float* out = (float*)d_out;

  unsigned short* gB = (unsigned short*)d_ws;   // 736*256*2 = 368 KB scratch
  const int B = in_sizes[1];                    // number of nodes (indices count)

  prep_B_kernel<<<NCOLP, KPAD, 0, stream>>>(U2, U3, gB);
  symcon_main_kernel<<<B, 256, 0, stream>>>(x, idx, W1, W2, W3, U1, gB, out);
}
